// Self_attention_pixel_wise_46909632807722
// MI455X (gfx1250) — matrix operand order repaired and checked by reading, not compiled, not run on hardware
//
#include <hip/hip_runtime.h>
#include <hip/hip_bf16.h>

typedef __attribute__((ext_vector_type(16))) __bf16 bf16x16;
typedef __attribute__((ext_vector_type(8)))  float  f32x8;

union Frag { bf16x16 v; uint4 q[2]; };
union F8   { f32x8 v; float f[8]; };

__device__ __forceinline__ unsigned short f2bf(float f) {
  unsigned u = __float_as_uint(f);
  u += 0x7FFFu + ((u >> 16) & 1u);          // round-to-nearest-even
  return (unsigned short)(u >> 16);
}

#define WMMA_BF16(a, b, c) \
  __builtin_amdgcn_wmma_f32_16x16x32_bf16(false, (a), false, (b), (short)0, (c), false, false)

// ---------------------------------------------------------------------------
// Kernel 0: transpose + bf16-convert the four weight matrices.
//   wk/wq/wv [256][128] f32  -> wT [128][256] bf16  (Bᵀ for x @ w)
//   wo       [128][256] f32  -> woT [256][128] bf16 (Bᵀ for o @ wo)
// ---------------------------------------------------------------------------
__global__ void prep_weights(const float* __restrict__ wk, const float* __restrict__ wq,
                             const float* __restrict__ wv, const float* __restrict__ wo,
                             unsigned short* __restrict__ wkT, unsigned short* __restrict__ wqT,
                             unsigned short* __restrict__ wvT, unsigned short* __restrict__ woT) {
  int i = blockIdx.x * blockDim.x + threadIdx.x;
  if (i < 128 * 256) {
    int d = i >> 8;      // 0..127
    int c = i & 255;     // 0..255
    wkT[d * 256 + c] = f2bf(wk[c * 128 + d]);
    wqT[d * 256 + c] = f2bf(wq[c * 128 + d]);
    wvT[d * 256 + c] = f2bf(wv[c * 128 + d]);
    woT[c * 128 + d] = f2bf(wo[d * 256 + c]);
  }
}

// ---------------------------------------------------------------------------
// Kernel 1: QKV projection.  One wave (32 threads) per 16 pixels.
//   K,Q stored row-major bf16 [B][4096][128];  V stored transposed [B][128][4096].
// ---------------------------------------------------------------------------
__global__ __launch_bounds__(32) void proj_kernel(
    const float* __restrict__ x,
    const unsigned short* __restrict__ wkT, const unsigned short* __restrict__ wqT,
    const unsigned short* __restrict__ wvT,
    const float* __restrict__ bk, const float* __restrict__ bq, const float* __restrict__ bv,
    unsigned short* __restrict__ Kb, unsigned short* __restrict__ Qb,
    unsigned short* __restrict__ VT) {
  __shared__ __align__(16) unsigned short xs[16 * 256];
  const int lane = threadIdx.x;
  const int hl = lane & 15;
  const int hi = lane >> 4;               // 0 = lanes 0-15, 1 = lanes 16-31
  const int bid = blockIdx.x;
  const int b  = bid >> 8;                // 256 row-blocks per batch
  const int n0 = (bid & 255) << 4;

  // Stage 16 rows of x (4096 contiguous floats) into LDS as bf16.
  const float* xrow = x + ((size_t)(b * 4096 + n0)) * 256;
  #pragma unroll 4
  for (int i = lane; i < 1024; i += 32) {
    float4 f = ((const float4*)xrow)[i];
    unsigned long long p = (unsigned long long)f2bf(f.x)
                         | ((unsigned long long)f2bf(f.y) << 16)
                         | ((unsigned long long)f2bf(f.z) << 32)
                         | ((unsigned long long)f2bf(f.w) << 48);
    ((unsigned long long*)xs)[i] = p;
  }
  __syncthreads();

  // A-fragments: 8 K-tiles of 32 over C=256.
  // ISA 16-bit A layout: lanes<16 hold K {0..7,16..23}, lanes>=16 hold {8..15,24..31}.
  Frag a[8];
  #pragma unroll
  for (int kt = 0; kt < 8; ++kt) {
    const unsigned short* p = xs + hl * 256 + kt * 32 + hi * 8;
    a[kt].q[0] = *(const uint4*)p;
    a[kt].q[1] = *(const uint4*)(p + 16);
  }

  #pragma unroll 1
  for (int o = 0; o < 3; ++o) {
    const unsigned short* wT   = (o == 0) ? wkT : ((o == 1) ? wqT : wvT);
    const float*          bias = (o == 0) ? bk  : ((o == 1) ? bq  : bv);
    F8 acc[8];
    #pragma unroll
    for (int dt = 0; dt < 8; ++dt) {
      f32x8 z = {0.f, 0.f, 0.f, 0.f, 0.f, 0.f, 0.f, 0.f};
      acc[dt].v = z;
    }
    #pragma unroll
    for (int kt = 0; kt < 8; ++kt) {
      #pragma unroll
      for (int dt = 0; dt < 8; ++dt) {
        // B-fragment from Bᵀ (=wT) row-major: column = 16*dt + hl, K {8hi..8hi+7, 16+8hi..23+8hi}.
        Frag bw;
        const unsigned short* p = wT + (dt * 16 + hl) * 256 + kt * 32 + hi * 8;
        bw.q[0] = *(const uint4*)p;
        bw.q[1] = *(const uint4*)(p + 16);
        acc[dt].v = WMMA_BF16(a[kt].v, bw.v, acc[dt].v);
      }
    }
    #pragma unroll
    for (int dt = 0; dt < 8; ++dt) {
      const int d  = dt * 16 + hl;
      const float bb = bias[d];
      #pragma unroll
      for (int r = 0; r < 8; ++r) {
        float val = acc[dt].f[r] + bb;       // D layout: row M = r + 8*hi, col N = hl
        int n = n0 + r + hi * 8;
        if (o == 2)
          VT[((size_t)(b * 128 + d)) * 4096 + n] = f2bf(val);
        else {
          unsigned short* dst = (o == 0) ? Kb : Qb;
          dst[((size_t)(b * 4096 + n)) * 128 + d] = f2bf(val);
        }
      }
    }
  }
}

// ---------------------------------------------------------------------------
// Kernel 2: fused flash attention + output projection + residual.
// One wave per 16 query rows; streams m in blocks of 32.
// ---------------------------------------------------------------------------
__global__ __launch_bounds__(32) void attn_kernel(
    const unsigned short* __restrict__ Kb, const unsigned short* __restrict__ Qb,
    const unsigned short* __restrict__ VT, const unsigned short* __restrict__ woT,
    const float* __restrict__ bo, const float* __restrict__ x,
    float* __restrict__ out) {
  __shared__ __align__(16) unsigned short pbuf[16 * 32];   // P tile relayout (D -> A)
  __shared__ __align__(16) unsigned short obuf[16 * 128];  // o tile relayout for out-proj
  const int lane = threadIdx.x;
  const int hl = lane & 15;
  const int hi = lane >> 4;
  const int bid = blockIdx.x;
  const int b  = bid >> 8;
  const int n0 = (bid & 255) << 4;

  // K A-fragments: 16 rows x 128 d, 4 K-tiles of 32 (kept resident all loop).
  Frag ak[4];
  const unsigned short* kp = Kb + ((size_t)(b * 4096 + n0 + hl)) * 128;
  #pragma unroll
  for (int kt = 0; kt < 4; ++kt) {
    const unsigned short* p = kp + kt * 32 + hi * 8;
    ak[kt].q[0] = *(const uint4*)p;
    ak[kt].q[1] = *(const uint4*)(p + 16);
  }

  F8 oacc[8];
  #pragma unroll
  for (int t = 0; t < 8; ++t) {
    f32x8 z = {0.f, 0.f, 0.f, 0.f, 0.f, 0.f, 0.f, 0.f};
    oacc[t].v = z;
  }
  float mrun[8], lrun[8];
  #pragma unroll
  for (int r = 0; r < 8; ++r) { mrun[r] = -3.0e38f; lrun[r] = 0.f; }

  const unsigned short* qbase = Qb + ((size_t)b * 4096) * 128;
  const unsigned short* vbase = VT + ((size_t)b * 128) * 4096;

  for (int j = 0; j < 128; ++j) {
    const int m0 = j * 32;
    F8 s0, s1;
    { f32x8 z = {0.f,0.f,0.f,0.f,0.f,0.f,0.f,0.f}; s0.v = z; s1.v = z; }
    #pragma unroll
    for (int kt = 0; kt < 4; ++kt) {
      Frag bq0, bq1;   // Bᵀ = Q row-major: column m = m0 + hl (+16)
      const unsigned short* p0 = qbase + (size_t)(m0 + hl) * 128 + kt * 32 + hi * 8;
      bq0.q[0] = *(const uint4*)p0;
      bq0.q[1] = *(const uint4*)(p0 + 16);
      const unsigned short* p1 = p0 + 16 * 128;
      bq1.q[0] = *(const uint4*)p1;
      bq1.q[1] = *(const uint4*)(p1 + 16);
      s0.v = WMMA_BF16(ak[kt].v, bq0.v, s0.v);
      s1.v = WMMA_BF16(ak[kt].v, bq1.v, s1.v);
    }

    // Online softmax over 32 columns. Rows live on (vgpr r, lane-half):
    // lanes<16 -> row r, lanes>=16 -> row r+8.  Half-wave reductions via shfl_xor.
    float scale[8];
    #pragma unroll
    for (int r = 0; r < 8; ++r) {
      float tv = fmaxf(s0.f[r], s1.f[r]);
      tv = fmaxf(tv, __shfl_xor(tv, 1));
      tv = fmaxf(tv, __shfl_xor(tv, 2));
      tv = fmaxf(tv, __shfl_xor(tv, 4));
      tv = fmaxf(tv, __shfl_xor(tv, 8));
      float mnew = fmaxf(mrun[r], tv);
      scale[r] = __expf(mrun[r] - mnew);
      mrun[r] = mnew;
      s0.f[r] = __expf(s0.f[r] - mnew);
      s1.f[r] = __expf(s1.f[r] - mnew);
      float ssum = s0.f[r] + s1.f[r];
      ssum += __shfl_xor(ssum, 1);
      ssum += __shfl_xor(ssum, 2);
      ssum += __shfl_xor(ssum, 4);
      ssum += __shfl_xor(ssum, 8);
      lrun[r] = lrun[r] * scale[r] + ssum;
    }
    #pragma unroll
    for (int t = 0; t < 8; ++t) {
      #pragma unroll
      for (int r = 0; r < 8; ++r) oacc[t].f[r] *= scale[r];
    }

    // Relayout P (D-layout) -> A-layout via a 1KB LDS round trip.
    #pragma unroll
    for (int r = 0; r < 8; ++r) {
      int row = r + hi * 8;
      pbuf[row * 32 + hl]      = f2bf(s0.f[r]);
      pbuf[row * 32 + 16 + hl] = f2bf(s1.f[r]);
    }
    __syncthreads();
    Frag pa;
    const unsigned short* pp = pbuf + hl * 32 + hi * 8;
    pa.q[0] = *(const uint4*)pp;
    pa.q[1] = *(const uint4*)(pp + 16);
    __syncthreads();

    // oacc += P @ V  (B operand = Vᵀ row-major, 8 d-tiles of 16)
    #pragma unroll
    for (int t = 0; t < 8; ++t) {
      Frag bv_;
      const unsigned short* vp = vbase + (size_t)(t * 16 + hl) * 4096 + m0 + hi * 8;
      bv_.q[0] = *(const uint4*)vp;
      bv_.q[1] = *(const uint4*)(vp + 16);
      oacc[t].v = WMMA_BF16(pa.v, bv_.v, oacc[t].v);
    }
  }

  // Normalize by running sum.
  #pragma unroll
  for (int r = 0; r < 8; ++r) {
    float inv = 1.0f / lrun[r];
    #pragma unroll
    for (int t = 0; t < 8; ++t) oacc[t].f[r] *= inv;
  }

  // Stage o (16x128) to LDS as bf16, reload as A-fragments for the out-proj.
  #pragma unroll
  for (int t = 0; t < 8; ++t) {
    #pragma unroll
    for (int r = 0; r < 8; ++r)
      obuf[(r + hi * 8) * 128 + t * 16 + hl] = f2bf(oacc[t].f[r]);
  }
  __syncthreads();

  Frag ao[4];
  #pragma unroll
  for (int kt = 0; kt < 4; ++kt) {
    const unsigned short* p = obuf + hl * 128 + kt * 32 + hi * 8;
    ao[kt].q[0] = *(const uint4*)p;
    ao[kt].q[1] = *(const uint4*)(p + 16);
  }

  const float* xrow = x   + ((size_t)(b * 4096 + n0)) * 256;
  float*       orow = out + ((size_t)(b * 4096 + n0)) * 256;
  #pragma unroll 1
  for (int ct = 0; ct < 16; ++ct) {
    F8 facc;
    { f32x8 z = {0.f,0.f,0.f,0.f,0.f,0.f,0.f,0.f}; facc.v = z; }
    #pragma unroll
    for (int kt = 0; kt < 4; ++kt) {
      Frag bw;   // Bᵀ = woT [256][128]
      const unsigned short* p = woT + (ct * 16 + hl) * 128 + kt * 32 + hi * 8;
      bw.q[0] = *(const uint4*)p;
      bw.q[1] = *(const uint4*)(p + 16);
      facc.v = WMMA_BF16(ao[kt].v, bw.v, facc.v);
    }
    const int c = ct * 16 + hl;
    const float bb = bo[c];
    #pragma unroll
    for (int r = 0; r < 8; ++r) {
      int n = r + hi * 8;
      orow[n * 256 + c] = facc.f[r] + bb + xrow[n * 256 + c];   // bias + residual
    }
  }
}

// ---------------------------------------------------------------------------
extern "C" void kernel_launch(void* const* d_in, const int* in_sizes, int n_in,
                              void* d_out, int out_size, void* d_ws, size_t ws_size,
                              hipStream_t stream) {
  const float* x  = (const float*)d_in[0];
  const float* wk = (const float*)d_in[1];
  const float* bk = (const float*)d_in[2];
  const float* wq = (const float*)d_in[3];
  const float* bq = (const float*)d_in[4];
  const float* wv = (const float*)d_in[5];
  const float* bv = (const float*)d_in[6];
  const float* wo = (const float*)d_in[7];
  const float* bo = (const float*)d_in[8];
  float* out = (float*)d_out;

  // Workspace layout (bf16 as raw u16):
  //   Kb  [4][4096][128]   4 MB
  //   Qb  [4][4096][128]   4 MB
  //   VT  [4][128][4096]   4 MB
  //   wkT/wqT/wvT [128][256], woT [256][128]   64 KB each
  unsigned short* Kb  = (unsigned short*)d_ws;
  unsigned short* Qb  = Kb  + (size_t)4 * 4096 * 128;
  unsigned short* VT  = Qb  + (size_t)4 * 4096 * 128;
  unsigned short* wkT = VT  + (size_t)4 * 4096 * 128;
  unsigned short* wqT = wkT + 128 * 256;
  unsigned short* wvT = wqT + 128 * 256;
  unsigned short* woT = wvT + 128 * 256;

  prep_weights<<<128, 256, 0, stream>>>(wk, wq, wv, wo, wkT, wqT, wvT, woT);
  proj_kernel<<<1024, 32, 0, stream>>>(x, wkT, wqT, wvT, bk, bq, bv, Kb, Qb, VT);
  attn_kernel<<<1024, 32, 0, stream>>>(Kb, Qb, VT, woT, bo, x, out);
}
